// Weighted_Dist_Displacement_UDF_10376640987892
// MI455X (gfx1250) — matrix operand order repaired and checked by reading, not compiled, not run on hardware
//
#include <hip/hip_runtime.h>
#include <hip/hip_bf16.h>
#include <stdint.h>

typedef _Float16 half_t;
typedef __attribute__((ext_vector_type(16))) _Float16 v16h;
typedef __attribute__((ext_vector_type(8)))  float    v8f;

union Frag16 { v16h v; uint32_t u[8]; half_t h[16]; };

#define LRELU_SLOPE 0.2f

// ---------------------------------------------------------------------------
// CDNA5 async global->LDS staging (ASYNCcnt path), with sync fallback.
// Builtin signature (from hipcc diagnostic): first param is
//   int __attribute__((vector_size(16))) __device__ *   (addrspace(1))
// ---------------------------------------------------------------------------
#if defined(__AMDGCN__) && __has_builtin(__builtin_amdgcn_global_load_async_to_lds_b128)
#define HAVE_ASYNC_LDS 1
typedef int b128_t __attribute__((vector_size(16)));
typedef b128_t __attribute__((address_space(1))) * gas_b128_ptr;
typedef b128_t __attribute__((address_space(3))) * las_b128_ptr;
#endif

__device__ __forceinline__ void copy16_to_lds(const void* g, void* l)
{
#ifdef HAVE_ASYNC_LDS
    __builtin_amdgcn_global_load_async_to_lds_b128(
        (gas_b128_ptr)g, (las_b128_ptr)l, 0, 0);
#else
    *(uint4*)l = *(const uint4*)g;
#endif
}

__device__ __forceinline__ void async_wait0()
{
#ifdef HAVE_ASYNC_LDS
#if __has_builtin(__builtin_amdgcn_s_wait_asynccnt)
    __builtin_amdgcn_s_wait_asynccnt(0);
#else
    asm volatile("s_wait_asynccnt 0x0" ::: "memory");
#endif
#endif
}

// ---------------------------------------------------------------------------
// prep: fold conv bias + BN into per-channel scale/shift (y = dot*a + c);
// emit weights PANEL-MAJOR f16: Wp[ks][n][32] with k = ks*32..ks*32+31,
// so GEMM weight staging is a contiguous async B128 copy.
// ---------------------------------------------------------------------------
__global__ void prep_layer(const float* __restrict__ w, const float* __restrict__ b,
                           const float* __restrict__ gamma, const float* __restrict__ beta,
                           const float* __restrict__ mean, const float* __restrict__ var,
                           int cin, int cinp, int cout, int coutp,
                           half_t* __restrict__ Wp, float* __restrict__ aS,
                           float* __restrict__ cS, int hasbn)
{
    int idx = blockIdx.x * blockDim.x + threadIdx.x;
    if (idx < coutp) {
        float s = 1.f, sh = 0.f;
        if (idx < cout) {
            if (hasbn) {
                s  = gamma[idx] * rsqrtf(var[idx] + 1e-5f);
                sh = b[idx] * s + beta[idx] - mean[idx] * s;
            } else {
                s = 1.f; sh = b[idx];
            }
        }
        aS[idx] = s; cS[idx] = sh;
    }
    if (idx < cinp * coutp) {
        int n = idx / cinp, kg = idx % cinp;
        int ks = kg >> 5, k = kg & 31;
        half_t v = (half_t)0.f;
        if (n < cout && kg < cin) v = (half_t)w[n * cin + kg];
        Wp[((size_t)ks * coutp + n) * 32 + k] = v;
    }
}

// ---------------------------------------------------------------------------
// Ball query (first-K-in-index-order, d2 < r^2) fused with feature build:
//   X0 (M x 32 f16): [local(3), q(3), 0...]   LOC (M x 4 f32): [local(3), |local|]
// Missing neighbors follow reference semantics: knn=0 -> local=q.
// ---------------------------------------------------------------------------
__global__ __launch_bounds__(256) void ball_build(
    const float* __restrict__ pcd, const float* __restrict__ qp,
    half_t* __restrict__ X0, float* __restrict__ LOC)
{
    const int Nn = 4096, Qn = 4096, Kk = 10;
    const float r2 = 0.02f * 0.02f;
    int b = blockIdx.x >> 4;
    int q = ((blockIdx.x & 15) << 8) + threadIdx.x;
    float qx = qp[((size_t)b * 3 + 0) * Qn + q];
    float qy = qp[((size_t)b * 3 + 1) * Qn + q];
    float qz = qp[((size_t)b * 3 + 2) * Qn + q];
    __shared__ float sp[256 * 3];
    int cnt = 0;
    size_t row0 = ((size_t)b * Qn + q) * Kk;
    for (int t0 = 0; t0 < Nn; t0 += 256) {
        __syncthreads();
        for (int i = threadIdx.x; i < 256 * 3; i += 256)
            sp[i] = pcd[((size_t)b * Nn + t0) * 3 + i];
        __syncthreads();
        if (cnt < Kk) {
            for (int i = 0; i < 256 && cnt < Kk; i++) {
                float dx = qx - sp[i * 3 + 0];
                float dy = qy - sp[i * 3 + 1];
                float dz = qz - sp[i * 3 + 2];
                float d2 = dx * dx + dy * dy + dz * dz;
                if (d2 < r2) {
                    size_t r = row0 + cnt;
                    half_t* xr = X0 + r * 32;
                    xr[0] = (half_t)dx; xr[1] = (half_t)dy; xr[2] = (half_t)dz;
                    xr[3] = (half_t)qx; xr[4] = (half_t)qy; xr[5] = (half_t)qz;
                    #pragma unroll
                    for (int c = 6; c < 32; c++) xr[c] = (half_t)0.f;
                    float* lr = LOC + r * 4;
                    lr[0] = dx; lr[1] = dy; lr[2] = dz; lr[3] = sqrtf(d2);
                    cnt++;
                }
            }
        }
    }
    for (; cnt < Kk; cnt++) {
        size_t r = row0 + cnt;
        half_t* xr = X0 + r * 32;
        xr[0] = (half_t)qx; xr[1] = (half_t)qy; xr[2] = (half_t)qz;
        xr[3] = (half_t)qx; xr[4] = (half_t)qy; xr[5] = (half_t)qz;
        for (int c = 6; c < 32; c++) xr[c] = (half_t)0.f;
        float* lr = LOC + r * 4;
        lr[0] = qx; lr[1] = qy; lr[2] = qz; lr[3] = sqrtf(qx*qx + qy*qy + qz*qz);
    }
}

// ---------------------------------------------------------------------------
// WMMA GEMM layer: Y(M x COUTP) = act( X(M x CINP) * W^T * a + c )
// 256 threads = 8 wave32 waves; block tile = 128 rows; wave owns 16 rows and
// all COUTP/16 n-tiles. Double-buffered async global->LDS staging overlapped
// with v_wmma_f32_16x16x32_f16; LDS rows padded to 40 halves (16B aligned,
// bank-conflict-free: dword stride 20 -> 16 distinct banks).
// Fragment layouts per CDNA5 ISA 7.12.2.
// ---------------------------------------------------------------------------
template<int CINP, int COUTP, bool ACT, bool OUTF32>
__global__ __launch_bounds__(256) void gemm_ws(
    const half_t* __restrict__ X, const half_t* __restrict__ Wp,
    const float* __restrict__ aS, const float* __restrict__ cS,
    half_t* __restrict__ Yh, float* __restrict__ Yf)
{
    static_assert(CINP % 32 == 0 && COUTP % 16 == 0, "pad dims");
    constexpr int NT     = COUTP / 16;
    constexpr int KSTEPS = CINP / 32;
    constexpr int XROW   = 40;                     // halves; 80B rows, 16B aligned
    __shared__ half_t sX[2][128 * XROW];
    __shared__ half_t sW[2][COUTP * XROW];
    const int tid  = threadIdx.x;
    const int lane = tid & 31;
    const int wave = tid >> 5;
    const int m0   = blockIdx.x * 128;

    auto stage = [&](int ks, int buf) {
        // activation tile: 128 rows x 64B, four 16B chunks per row
        for (int idx = tid; idx < 128 * 4; idx += 256) {
            int r = idx >> 2, c = idx & 3;
            copy16_to_lds(X + (size_t)(m0 + r) * CINP + ks * 32 + c * 8,
                          &sX[buf][r * XROW + c * 8]);
        }
        // weight K-panel (panel-major in global): COUTP rows x 64B
        for (int idx = tid; idx < COUTP * 4; idx += 256) {
            int n = idx >> 2, c = idx & 3;
            copy16_to_lds(Wp + ((size_t)ks * COUTP + n) * 32 + c * 8,
                          &sW[buf][n * XROW + c * 8]);
        }
    };

    v8f acc[NT];
    #pragma unroll
    for (int t = 0; t < NT; t++)
        #pragma unroll
        for (int j = 0; j < 8; j++) acc[t][j] = 0.f;

    stage(0, 0);
    async_wait0();
    __syncthreads();

    int cur = 0;
    for (int ks = 0; ks < KSTEPS; ks++) {
        if (ks + 1 < KSTEPS) stage(ks + 1, cur ^ 1);   // prefetch next panel (async)

        // A fragment: row = wave*16 + (lane&15); K halves per ISA A 16x32 layout
        Frag16 a;
        {
            int m = wave * 16 + (lane & 15);
            const uint32_t* row = (const uint32_t*)(&sX[cur][m * XROW]);
            const int kb0 = (lane < 16) ? 0 : 4;       // dword index: K {0..7 | 8..15}
            const int kb1 = (lane < 16) ? 8 : 12;      // dword index: K {16..23 | 24..31}
            #pragma unroll
            for (int j = 0; j < 4; j++) a.u[j]     = row[kb0 + j];
            #pragma unroll
            for (int j = 0; j < 4; j++) a.u[4 + j] = row[kb1 + j];
        }
        #pragma unroll
        for (int nt = 0; nt < NT; nt++) {
            // B fragment: col = nt*16 + (lane&15); K halves match A layout
            Frag16 bfr;
            int n = nt * 16 + (lane & 15);
            const uint32_t* col = (const uint32_t*)(&sW[cur][n * XROW]);
            const int kb = (lane < 16) ? 0 : 8;
            #pragma unroll
            for (int j = 0; j < 8; j++) bfr.u[j] = col[(kb >> 1) + (j & 3) + ((j >> 2) << 3)];
            acc[nt] = __builtin_amdgcn_wmma_f32_16x16x32_f16(
                false, a.v, false, bfr.v, (short)0, acc[nt], false, false);
        }

        if (ks + 1 < KSTEPS) {
            async_wait0();                             // next panel landed in LDS
            __syncthreads();
        }
        cur ^= 1;
    }

    // epilogue: scale/shift (+leaky); C/D layout: row = v + (lane<16?0:8)
    #pragma unroll
    for (int nt = 0; nt < NT; nt++) {
        int n = nt * 16 + (lane & 15);
        float s = aS[n], sh = cS[n];
        int mb = m0 + wave * 16 + ((lane < 16) ? 0 : 8);
        #pragma unroll
        for (int v = 0; v < 8; v++) {
            float y = acc[nt][v] * s + sh;
            if (ACT) y = (y > 0.f) ? y : LRELU_SLOPE * y;
            size_t off = (size_t)(mb + v) * COUTP + n;
            if constexpr (OUTF32) Yf[off] = y; else Yh[off] = (half_t)y;
        }
    }
}

// max over the K=10 neighbor slots
__global__ void patch_max(const half_t* __restrict__ FEAT, half_t* __restrict__ PATCH, int BQ)
{
    int idx = blockIdx.x * 256 + threadIdx.x;
    if (idx >= BQ * 128) return;
    int c = idx & 127, bq = idx >> 7;
    size_t base = (size_t)bq * 10 * 128 + c;
    float m = (float)FEAT[base];
    #pragma unroll
    for (int k = 1; k < 10; k++) m = fmaxf(m, (float)FEAT[base + (size_t)k * 128]);
    PATCH[idx] = (half_t)m;
}

// cv2 = [cv1(6), k_dist(1), feat(128), patch(128), pad(25)] -> M x 288 f16
__global__ void assemble(const half_t* __restrict__ X0, const float* __restrict__ LOC,
                         const half_t* __restrict__ FEAT, const half_t* __restrict__ PATCH,
                         half_t* __restrict__ XA, int M)
{
    size_t idx = (size_t)blockIdx.x * 256 + threadIdx.x;
    if (idx >= (size_t)M * 288) return;
    int c = (int)(idx % 288);
    size_t row = idx / 288;
    half_t v;
    if      (c < 6)   v = X0[row * 32 + c];
    else if (c == 6)  v = (half_t)LOC[row * 4 + 3];
    else if (c < 135) v = FEAT[row * 128 + (c - 7)];
    else if (c < 263) v = PATCH[(row / 10) * 128 + (c - 135)];
    else              v = (half_t)0.f;
    XA[idx] = v;
}

// softmax over k of att ch0; vec = sum_k w*(local - disp); udf = |vec|
__global__ void finalize(const float* __restrict__ ATT, const float* __restrict__ LOC,
                         float* __restrict__ out, int BQ)
{
    int bq = blockIdx.x * 256 + threadIdx.x;
    if (bq >= BQ) return;
    size_t base = (size_t)bq * 10;
    float lg[10], mx = -3.0e38f;
    #pragma unroll
    for (int k = 0; k < 10; k++) { lg[k] = ATT[(base + k) * 16 + 0]; mx = fmaxf(mx, lg[k]); }
    float s = 0.f;
    #pragma unroll
    for (int k = 0; k < 10; k++) { lg[k] = expf(lg[k] - mx); s += lg[k]; }
    float inv = 1.f / s;
    float vx = 0.f, vy = 0.f, vz = 0.f;
    #pragma unroll
    for (int k = 0; k < 10; k++) {
        const float* l = LOC + (base + k) * 4;
        const float* a = ATT + (base + k) * 16;
        float w = lg[k] * inv;
        vx += w * (l[0] - a[1]);
        vy += w * (l[1] - a[2]);
        vz += w * (l[2] - a[3]);
    }
    out[bq] = sqrtf(vx * vx + vy * vy + vz * vz);
}

// ---------------------------------------------------------------------------
extern "C" void kernel_launch(void* const* d_in, const int* in_sizes, int n_in,
                              void* d_out, int out_size, void* d_ws, size_t ws_size,
                              hipStream_t stream)
{
    (void)in_sizes; (void)n_in; (void)out_size; (void)ws_size;
    const int B = 4, Q = 4096, Kk = 10;
    const int M  = B * Q * Kk;          // 163840 (multiple of 128)
    const int BQ = B * Q;

    int ii = 0;
    const float* pcd = (const float*)d_in[ii++];
    const float* qp  = (const float*)d_in[ii++];
    const float *W[8], *Bi[8], *G[8], *Be[8], *Me[8], *Va[8];
    for (int g = 0; g < 2; g++) {
        for (int l = 0; l < 4; l++) {
            int li = g * 4 + l;
            W[li]  = (const float*)d_in[ii++];
            Bi[li] = (const float*)d_in[ii++];
            if (l < 3) {
                G[li]  = (const float*)d_in[ii++];
                Be[li] = (const float*)d_in[ii++];
                Me[li] = (const float*)d_in[ii++];
                Va[li] = (const float*)d_in[ii++];
            } else { G[li] = Be[li] = Me[li] = Va[li] = nullptr; }
        }
    }

    uint8_t* p = (uint8_t*)d_ws;
    auto carve = [&](size_t bytes) -> void* {
        void* r = (void*)p; p += (bytes + 255) & ~(size_t)255; return r;
    };
    half_t* X0    = (half_t*)carve((size_t)M * 32  * 2);
    float*  LOC   = (float*) carve((size_t)M * 4   * 4);
    half_t* BUFB  = (half_t*)carve((size_t)M * 256 * 2);
    half_t* BUFC  = (half_t*)carve((size_t)M * 128 * 2);
    half_t* XA    = (half_t*)carve((size_t)M * 288 * 2);
    half_t* PATCH = (half_t*)carve((size_t)BQ * 128 * 2);
    float*  ATT   = (float*) carve((size_t)M * 16  * 4);

    const int cin[8]   = {6, 64, 128, 128, 263, 256, 128, 32};
    const int cinp[8]  = {32, 64, 128, 128, 288, 256, 128, 32};
    const int cout[8]  = {64, 128, 128, 128, 256, 128, 32, 4};
    const int coutp[8] = {64, 128, 128, 128, 256, 128, 32, 16};
    half_t* Wt[8]; float* aS[8]; float* cS[8];
    for (int l = 0; l < 8; l++) {
        Wt[l] = (half_t*)carve((size_t)cinp[l] * coutp[l] * 2);
        aS[l] = (float*) carve((size_t)coutp[l] * 4);
        cS[l] = (float*) carve((size_t)coutp[l] * 4);
    }

    for (int l = 0; l < 8; l++) {
        int total = cinp[l] * coutp[l];
        prep_layer<<<(total + 255) / 256, 256, 0, stream>>>(
            W[l], Bi[l], G[l], Be[l], Me[l], Va[l],
            cin[l], cinp[l], cout[l], coutp[l],
            Wt[l], aS[l], cS[l], G[l] != nullptr ? 1 : 0);
    }

    ball_build<<<B * (Q / 256), 256, 0, stream>>>(pcd, qp, X0, LOC);

    const int GB = M / 128;
    // pf MLP: 6->64->128->128->128 (last layer: no BN/act)
    gemm_ws<32, 64, true, false><<<GB, 256, 0, stream>>>(X0,   Wt[0], aS[0], cS[0], BUFB, nullptr);
    gemm_ws<64, 128, true, false><<<GB, 256, 0, stream>>>(BUFB, Wt[1], aS[1], cS[1], BUFC, nullptr);
    gemm_ws<128, 128, true, false><<<GB, 256, 0, stream>>>(BUFC, Wt[2], aS[2], cS[2], BUFB, nullptr);
    gemm_ws<128, 128, false, false><<<GB, 256, 0, stream>>>(BUFB, Wt[3], aS[3], cS[3], BUFC, nullptr); // FEAT

    patch_max<<<(BQ * 128 + 255) / 256, 256, 0, stream>>>(BUFC, PATCH, BQ);
    {
        size_t tot = (size_t)M * 288;
        assemble<<<(int)((tot + 255) / 256), 256, 0, stream>>>(X0, LOC, BUFC, PATCH, XA, M);
    }

    // attn MLP: 263(->288)->256->128->32->4(->16), last layer f32 out
    gemm_ws<288, 256, true, false><<<GB, 256, 0, stream>>>(XA,   Wt[4], aS[4], cS[4], BUFB, nullptr);
    gemm_ws<256, 128, true, false><<<GB, 256, 0, stream>>>(BUFB, Wt[5], aS[5], cS[5], BUFC, nullptr);
    gemm_ws<128, 32, true, false><<<GB, 256, 0, stream>>>(BUFC, Wt[6], aS[6], cS[6], BUFB, nullptr);
    gemm_ws<32, 16, false, true><<<GB, 256, 0, stream>>>(BUFB, Wt[7], aS[7], cS[7], nullptr, ATT);

    finalize<<<(BQ + 255) / 256, 256, 0, stream>>>(ATT, LOC, (float*)d_out, BQ);
}
